// CombinedModel_75617194213650
// MI455X (gfx1250) — matrix operand order repaired and checked by reading, not compiled, not run on hardware
//
#include <hip/hip_runtime.h>
#include <math.h>

#define NN 50000
#define NE 800000

typedef __attribute__((ext_vector_type(16))) __bf16 v16bf;
typedef __attribute__((ext_vector_type(8)))  __bf16 v8bf;
typedef __attribute__((ext_vector_type(8)))  float  v8f;

__device__ __forceinline__ v8f v8f_zero() {
  v8f z;
#pragma unroll
  for (int i = 0; i < 8; ++i) z[i] = 0.0f;
  return z;
}

// ---- fragment loaders -------------------------------------------------------
// A fragment (16x32 bf16): lane&15 = row; lane>>4 selects K {0..7,16..23} vs
// {8..15,24..31}. fp32 source: convert in-register. bf16 source: raw b128 loads.
__device__ __forceinline__ v16bf load_a(const float* rowp, int k0, int hi) {
  v16bf a;
  const float* p0 = rowp + k0 + hi * 8;
#pragma unroll
  for (int j = 0; j < 8; ++j) {
    a[j]     = (__bf16)p0[j];
    a[j + 8] = (__bf16)p0[j + 16];
  }
  return a;
}

__device__ __forceinline__ v16bf load_a(const __bf16* rowp, int k0, int hi) {
  const __bf16* p0 = rowp + k0 + hi * 8;
  v8bf lo = *(const v8bf*)p0;
  v8bf hh = *(const v8bf*)(p0 + 16);
  return __builtin_shufflevector(lo, hh, 0, 1, 2, 3, 4, 5, 6, 7, 8, 9, 10, 11, 12,
                                 13, 14, 15);
}

// B fragment (32x16 bf16): lane&15 = N column; lane>>4 selects K {0..7,16..23} vs
// {8..15,24..31}
__device__ __forceinline__ v16bf load_b(const __bf16* W, int ldw, int k0, int nbase,
                                        int lane) {
  const __bf16* p = W + (size_t)(k0 + (lane >> 4) * 8) * ldw + nbase + (lane & 15);
  v16bf b;
#pragma unroll
  for (int j = 0; j < 8; ++j) {
    b[j]     = p[(size_t)j * ldw];
    b[j + 8] = p[(size_t)(j + 16) * ldw];
  }
  return b;
}

__device__ __forceinline__ v8f wmma_bf16(v16bf a, v16bf b, v8f c) {
  return __builtin_amdgcn_wmma_f32_16x16x32_bf16(false, a, false, b, (short)0, c,
                                                 false, false);
}

// Y(16x64) = Xtile(16x64) @ W[m](64x64 bf16) for m=0..NMAT-1, shared A loads
template <typename AT, int NMAT>
__device__ __forceinline__ void mm16x64(const AT* a_row,
                                        const __bf16* const (&Ws)[NMAT], int lane,
                                        v8f (&acc)[NMAT][4]) {
  int hi = lane >> 4;
#pragma unroll
  for (int m = 0; m < NMAT; ++m)
#pragma unroll
    for (int nt = 0; nt < 4; ++nt) acc[m][nt] = v8f_zero();
#pragma unroll
  for (int ks = 0; ks < 2; ++ks) {
    v16bf a = load_a(a_row, ks * 32, hi);
#pragma unroll
    for (int m = 0; m < NMAT; ++m)
#pragma unroll
      for (int nt = 0; nt < 4; ++nt)
        acc[m][nt] = wmma_bf16(a, load_b(Ws[m], 64, ks * 32, nt * 16, lane),
                               acc[m][nt]);
  }
}

// D layout: lane&15 = col within n-tile; VGPR i + 8*(lane>>4) = row
__device__ __forceinline__ void store_tile64(float* out, int row0,
                                             const v8f (&acc)[4], int lane) {
  int l = lane & 15, hi = lane >> 4;
#pragma unroll
  for (int nt = 0; nt < 4; ++nt)
#pragma unroll
    for (int i = 0; i < 8; ++i)
      out[(size_t)(row0 + hi * 8 + i) * 64 + nt * 16 + l] = acc[nt][i];
}

__device__ __forceinline__ void atomic_max_f32(float* addr, float v) {
  if (v >= 0.0f)
    atomicMax(reinterpret_cast<int*>(addr), __float_as_int(v));
  else
    atomicMin(reinterpret_cast<unsigned int*>(addr), __float_as_uint(v));
}

__device__ __forceinline__ float gelu_tanh(float x) {
  float x3 = x * x * x;
  float t = tanhf(0.7978845608028654f * (x + 0.044715f * x3));
  return 0.5f * x * (1.0f + t);
}

// ---------------- kernels ----------------

__global__ void k_init(float* m_n, float* m_e, float* den_n, float* den_e,
                       float* msg, float* pooled) {
  int stride = gridDim.x * blockDim.x;
  for (int i = blockIdx.x * blockDim.x + threadIdx.x; i < NN * 64; i += stride) {
    msg[i] = 0.0f;
    pooled[i] = 0.0f;
    if (i < NN * 4) {
      m_n[i] = -__builtin_inff();
      m_e[i] = -__builtin_inff();
      den_n[i] = 0.0f;
      den_e[i] = 0.0f;
    }
  }
}

__global__ void k_cvt(const float* src, __bf16* dst, int n) {
  int i = blockIdx.x * blockDim.x + threadIdx.x;
  if (i < n) dst[i] = (__bf16)src[i];
}

// W2 (64x40) -> padded 64x48 bf16 (zero columns 40..47)
__global__ void k_cvt_pad(const float* W2, __bf16* dst) {
  int i = blockIdx.x * blockDim.x + threadIdx.x;
  if (i < 64 * 48) {
    int r = i / 48, c = i - r * 48;
    dst[i] = (c < 40) ? (__bf16)W2[r * 40 + c] : (__bf16)0.0f;
  }
}

// q = nodes@Wq_n, kn = nodes@Wk_n, vn = nodes@Wv_n, ne = nodes@Wn_e
__global__ void __launch_bounds__(128) k_nodeproj(const float* nodes,
                                                  const __bf16* Wq, const __bf16* Wk,
                                                  const __bf16* Wv, const __bf16* Wn,
                                                  float* q, float* kn, float* vn,
                                                  float* ne) {
  int wid = threadIdx.x >> 5, lane = threadIdx.x & 31;
  int r0 = blockIdx.x * 16;
  int l = lane & 15;
  const __bf16* Wsel = (wid == 0) ? Wq : (wid == 1) ? Wk : (wid == 2) ? Wv : Wn;
  float* osel = (wid == 0) ? q : (wid == 1) ? kn : (wid == 2) ? vn : ne;
  const __bf16* const Ws[1] = {Wsel};
  v8f acc[1][4];
  mm16x64<float, 1>(nodes + (size_t)(r0 + l) * 64, Ws, lane, acc);
  store_tile64(osel, r0, acc[0], lane);
}

// per 16-edge tile: s_e from (edges@Wq_e)·(edges@Wk_e + ne[src]),
//                   s_n from q[tgt]·(edges@We_n + kn[src]); atomicMax maxima.
// Side effect: writes bf16 copy of edges (ebf) for downstream kernels.
__global__ void __launch_bounds__(128) k_edge_scores(
    const float* edges, const int* src, const int* tgt, const float* q,
    const float* kn, const float* ne, const __bf16* Wq_e, const __bf16* Wk_e,
    const __bf16* We_n, float* s_n, float* s_e, float* m_n, float* m_e,
    __bf16* ebf) {
  __shared__ float smA[4][1024];
  __shared__ float smB[4][1024];
  int wid = threadIdx.x >> 5, lane = threadIdx.x & 31;
  int l = lane & 15, hi = lane >> 4;
  int e0 = (blockIdx.x * 4 + wid) * 16;
  const float* a_row = edges + (size_t)(e0 + l) * 64;

  // bf16 copy of this tile (each lane covers its row's 32 elements)
  {
    __bf16* erow = ebf + (size_t)(e0 + l) * 64;
#pragma unroll
    for (int ks = 0; ks < 2; ++ks) {
      const float* p0 = a_row + ks * 32 + hi * 8;
      v8bf b0, b1;
#pragma unroll
      for (int j = 0; j < 8; ++j) {
        b0[j] = (__bf16)p0[j];
        b1[j] = (__bf16)p0[j + 16];
      }
      *(v8bf*)(erow + ks * 32 + hi * 8) = b0;
      *(v8bf*)(erow + ks * 32 + hi * 8 + 16) = b1;
    }
  }

  {
    const __bf16* const Ws[2] = {Wq_e, Wk_e};
    v8f acc[2][4];
    mm16x64<float, 2>(a_row, Ws, lane, acc);
    store_tile64(&smA[wid][0], 0, acc[0], lane);
    store_tile64(&smB[wid][0], 0, acc[1], lane);
  }
  __syncthreads();
  int e = e0 + l;
  int s = src[e], t = tgt[e];
  {
    const float* qe = &smA[wid][l * 64];
    const float* kp = &smB[wid][l * 64];
    const float* nrow = ne + (size_t)s * 64;
#pragma unroll
    for (int h2 = 0; h2 < 2; ++h2) {
      int h = hi * 2 + h2;
      float sum = 0.0f;
#pragma unroll
      for (int d0 = 0; d0 < 16; ++d0) {
        int dim = h * 16 + d0;
        sum += qe[dim] * (kp[dim] + nrow[dim]);
      }
      sum *= 0.25f;  // 1/sqrt(16)
      s_e[(size_t)e * 4 + h] = sum;
      atomic_max_f32(&m_e[(size_t)s * 4 + h], sum);
    }
  }
  __syncthreads();
  {
    const __bf16* const Ws[1] = {We_n};
    v8f acc[1][4];
    mm16x64<float, 1>(a_row, Ws, lane, acc);
    store_tile64(&smA[wid][0], 0, acc[0], lane);
  }
  __syncthreads();
  {
    const float* kp = &smA[wid][l * 64];
    const float* qrow = q + (size_t)t * 64;
    const float* krow = kn + (size_t)s * 64;
#pragma unroll
    for (int h2 = 0; h2 < 2; ++h2) {
      int h = hi * 2 + h2;
      float sum = 0.0f;
#pragma unroll
      for (int d0 = 0; d0 < 16; ++d0) {
        int dim = h * 16 + d0;
        sum += qrow[dim] * (kp[dim] + krow[dim]);
      }
      sum *= 0.25f;
      s_n[(size_t)e * 4 + h] = sum;
      atomic_max_f32(&m_n[(size_t)t * 4 + h], sum);
    }
  }
}

// ex = exp(s - max[seg]); store in place; accumulate denominators
__global__ void k_expsum(float* s_n, float* s_e, const int* src, const int* tgt,
                         const float* m_n, const float* m_e, float* den_n,
                         float* den_e) {
  int stride = gridDim.x * blockDim.x;
  const int total = NE * 4;
  for (int i = blockIdx.x * blockDim.x + threadIdx.x; i < total; i += stride) {
    int e = i >> 2, h = i & 3;
    int t = tgt[e], s = src[e];
    float exn = __expf(s_n[i] - m_n[(size_t)t * 4 + h]);
    s_n[i] = exn;
    atomicAdd(&den_n[(size_t)t * 4 + h], exn);
    float exe = __expf(s_e[i] - m_e[(size_t)s * 4 + h]);
    s_e[i] = exe;
    atomicAdd(&den_e[(size_t)s * 4 + h], exe);
  }
}

// recompute ve = edges@Wv_e per tile (from bf16 edges); scatter-add
// msg[tgt] += a_n*vn[src], pooled[src] += a_e*ve
__global__ void __launch_bounds__(256) k_aggregate(
    const __bf16* ebf, const int* src, const int* tgt, const float* vn,
    const __bf16* Wv_e, const float* exn, const float* exe, const float* den_n,
    const float* den_e, float* msg, float* pooled) {
  __shared__ float smV[8][1024];
  int wid = threadIdx.x >> 5, lane = threadIdx.x & 31;
  int l = lane & 15, hi = lane >> 4;
  int e0 = (blockIdx.x * 8 + wid) * 16;
  {
    const __bf16* const Ws[1] = {Wv_e};
    v8f acc[1][4];
    mm16x64<__bf16, 1>(ebf + (size_t)(e0 + l) * 64, Ws, lane, acc);
    store_tile64(&smV[wid][0], 0, acc[0], lane);
  }
  __syncthreads();
  int e = e0 + l, s = src[e], t = tgt[e];
  float an[2], ae[2];
#pragma unroll
  for (int h2 = 0; h2 < 2; ++h2) {
    int h = hi * 2 + h2;
    an[h2] = exn[(size_t)e * 4 + h] / (den_n[(size_t)t * 4 + h] + 1e-9f);
    ae[h2] = exe[(size_t)e * 4 + h] / (den_e[(size_t)s * 4 + h] + 1e-9f);
  }
  const float* vrow = vn + (size_t)s * 64;
#pragma unroll
  for (int j = 0; j < 32; ++j) {
    int d = hi * 32 + j;
    int h2 = j >> 4;
    atomicAdd(&msg[(size_t)t * 64 + d], an[h2] * vrow[d]);
    atomicAdd(&pooled[(size_t)s * 64 + d], ae[h2] * smV[wid][l * 64 + d]);
  }
}

// nn_bf = bf16(nodes + msg@Wo_n) ; po = pooled@Wo_e
__global__ void __launch_bounds__(64) k_outproj(const float* nodes, const float* msg,
                                                const float* pooled,
                                                const __bf16* Wo_n,
                                                const __bf16* Wo_e, __bf16* nnb,
                                                float* po) {
  int wid = threadIdx.x >> 5, lane = threadIdx.x & 31;
  int r0 = blockIdx.x * 16;
  int l = lane & 15, hi = lane >> 4;
  v8f acc[1][4];
  if (wid == 0) {
    const __bf16* const Ws[1] = {Wo_n};
    mm16x64<float, 1>(msg + (size_t)(r0 + l) * 64, Ws, lane, acc);
#pragma unroll
    for (int nt = 0; nt < 4; ++nt)
#pragma unroll
      for (int i = 0; i < 8; ++i) {
        int r = r0 + hi * 8 + i, c = nt * 16 + l;
        nnb[(size_t)r * 64 + c] =
            (__bf16)(nodes[(size_t)r * 64 + c] + acc[0][nt][i]);
      }
  } else {
    const __bf16* const Ws[1] = {Wo_e};
    mm16x64<float, 1>(pooled + (size_t)(r0 + l) * 64, Ws, lane, acc);
    store_tile64(po, r0, acc[0], lane);
  }
}

// out = (gelu([nn[src], nn[tgt], edges + po[src]] @ W1 + b1)) @ W2 + b2
__global__ void __launch_bounds__(256) k_classifier(
    const __bf16* ebf, const int* src, const int* tgt, const __bf16* nnb,
    const float* po, const __bf16* W1b, const float* b1, const __bf16* W2b,
    const float* b2, float* out) {
  __shared__ alignas(16) __bf16 smH[8][1024];
  int wid = threadIdx.x >> 5, lane = threadIdx.x & 31;
  int l = lane & 15, hi = lane >> 4;
  int e0 = (blockIdx.x * 8 + wid) * 16;
  int er = e0 + l;
  int sr = src[er], tr = tgt[er];
  const __bf16* r0p = nnb + (size_t)sr * 64;
  const __bf16* r1p = nnb + (size_t)tr * 64;
  const __bf16* r2p = ebf + (size_t)er * 64;
  const float* addp = po + (size_t)sr * 64;
  v8f acc[4];
#pragma unroll
  for (int nt = 0; nt < 4; ++nt) acc[nt] = v8f_zero();
  // segments 0,1: gathered new_nodes rows (bf16, L2-resident)
#pragma unroll
  for (int seg = 0; seg < 2; ++seg) {
    const __bf16* rp = seg ? r1p : r0p;
#pragma unroll
    for (int ks = 0; ks < 2; ++ks) {
      v16bf a = load_a(rp, ks * 32, hi);
      int k0 = seg * 64 + ks * 32;
#pragma unroll
      for (int nt = 0; nt < 4; ++nt)
        acc[nt] = wmma_bf16(a, load_b(W1b, 64, k0, nt * 16, lane), acc[nt]);
    }
  }
  // segment 2: new_edges = edges + po[src] formed on the fly
#pragma unroll
  for (int ks = 0; ks < 2; ++ks) {
    v16bf a;
    const __bf16* p0 = r2p + ks * 32 + hi * 8;
    const float* q0 = addp + ks * 32 + hi * 8;
#pragma unroll
    for (int j = 0; j < 8; ++j) {
      a[j]     = (__bf16)((float)p0[j] + q0[j]);
      a[j + 8] = (__bf16)((float)p0[j + 16] + q0[j + 16]);
    }
    int k0 = 128 + ks * 32;
#pragma unroll
    for (int nt = 0; nt < 4; ++nt)
      acc[nt] = wmma_bf16(a, load_b(W1b, 64, k0, nt * 16, lane), acc[nt]);
  }
  // epilogue: bias + GELU -> LDS (bf16)
#pragma unroll
  for (int nt = 0; nt < 4; ++nt)
#pragma unroll
    for (int i = 0; i < 8; ++i) {
      int col = nt * 16 + l;
      smH[wid][(hi * 8 + i) * 64 + col] = (__bf16)gelu_tanh(acc[nt][i] + b1[col]);
    }
  __syncthreads();
  const __bf16* a2row = &smH[wid][l * 64];
  v8f acc2[3];
#pragma unroll
  for (int nt = 0; nt < 3; ++nt) acc2[nt] = v8f_zero();
#pragma unroll
  for (int ks = 0; ks < 2; ++ks) {
    v16bf a = load_a(a2row, ks * 32, hi);
#pragma unroll
    for (int nt = 0; nt < 3; ++nt)
      acc2[nt] = wmma_bf16(a, load_b(W2b, 48, ks * 32, nt * 16, lane), acc2[nt]);
  }
#pragma unroll
  for (int nt = 0; nt < 3; ++nt)
#pragma unroll
    for (int i = 0; i < 8; ++i) {
      int n = nt * 16 + l;
      if (n < 40)
        out[(size_t)(e0 + hi * 8 + i) * 40 + n] = acc2[nt][i] + b2[n];
    }
}

// ---------------- host launcher ----------------

extern "C" void kernel_launch(void* const* d_in, const int* in_sizes, int n_in,
                              void* d_out, int out_size, void* d_ws, size_t ws_size,
                              hipStream_t stream) {
  (void)in_sizes; (void)n_in; (void)out_size; (void)ws_size;
  const float* nodes = (const float*)d_in[0];
  const float* edges = (const float*)d_in[1];
  const int* ei = (const int*)d_in[2];
  const int* src = ei;
  const int* tgt = ei + NE;
  const float* Wq_n = (const float*)d_in[3];
  const float* Wk_n = (const float*)d_in[4];
  const float* We_n = (const float*)d_in[5];
  const float* Wv_n = (const float*)d_in[6];
  const float* Wo_n = (const float*)d_in[7];
  const float* Wq_e = (const float*)d_in[8];
  const float* Wk_e = (const float*)d_in[9];
  const float* Wn_e = (const float*)d_in[10];
  const float* Wv_e = (const float*)d_in[11];
  const float* Wo_e = (const float*)d_in[12];
  const float* W1 = (const float*)d_in[13];
  const float* b1 = (const float*)d_in[14];
  const float* W2 = (const float*)d_in[15];
  const float* b2 = (const float*)d_in[16];
  float* out = (float*)d_out;

  float* ws = (float*)d_ws;
  size_t o = 0;
  float* q = ws + o;      o += (size_t)NN * 64;
  float* kn = ws + o;     o += (size_t)NN * 64;
  float* vn = ws + o;     o += (size_t)NN * 64;
  float* ne = ws + o;     o += (size_t)NN * 64;
  float* s_n = ws + o;    o += (size_t)NE * 4;
  float* s_e = ws + o;    o += (size_t)NE * 4;
  float* m_n = ws + o;    o += (size_t)NN * 4;
  float* m_e = ws + o;    o += (size_t)NN * 4;
  float* den_n = ws + o;  o += (size_t)NN * 4;
  float* den_e = ws + o;  o += (size_t)NN * 4;
  float* msg = ws + o;    o += (size_t)NN * 64;
  float* pooled = ws + o; o += (size_t)NN * 64;
  float* po = ws + o;     o += (size_t)NN * 64;

  __bf16* bbase = (__bf16*)(ws + o);  // byte offset divisible by 16
  size_t ob = 0;
  __bf16* ebf = bbase + ob;  ob += (size_t)NE * 64;
  __bf16* nnb = bbase + ob;  ob += (size_t)NN * 64;
  __bf16* wbf = bbase + ob;
  __bf16* wb_q_n = wbf + 0 * 4096;
  __bf16* wb_k_n = wbf + 1 * 4096;
  __bf16* wb_v_n = wbf + 2 * 4096;
  __bf16* wb_n_e = wbf + 3 * 4096;
  __bf16* wb_q_e = wbf + 4 * 4096;
  __bf16* wb_k_e = wbf + 5 * 4096;
  __bf16* wb_e_n = wbf + 6 * 4096;
  __bf16* wb_v_e = wbf + 7 * 4096;
  __bf16* wb_o_n = wbf + 8 * 4096;
  __bf16* wb_o_e = wbf + 9 * 4096;
  __bf16* wb_W1 = wbf + 10 * 4096;            // 192*64
  __bf16* wb_W2 = wbf + 10 * 4096 + 12288;    // 64*48 padded

  k_init<<<2048, 256, 0, stream>>>(m_n, m_e, den_n, den_e, msg, pooled);
  k_cvt<<<16, 256, 0, stream>>>(Wq_n, wb_q_n, 4096);
  k_cvt<<<16, 256, 0, stream>>>(Wk_n, wb_k_n, 4096);
  k_cvt<<<16, 256, 0, stream>>>(Wv_n, wb_v_n, 4096);
  k_cvt<<<16, 256, 0, stream>>>(Wn_e, wb_n_e, 4096);
  k_cvt<<<16, 256, 0, stream>>>(Wq_e, wb_q_e, 4096);
  k_cvt<<<16, 256, 0, stream>>>(Wk_e, wb_k_e, 4096);
  k_cvt<<<16, 256, 0, stream>>>(We_n, wb_e_n, 4096);
  k_cvt<<<16, 256, 0, stream>>>(Wv_e, wb_v_e, 4096);
  k_cvt<<<16, 256, 0, stream>>>(Wo_n, wb_o_n, 4096);
  k_cvt<<<16, 256, 0, stream>>>(Wo_e, wb_o_e, 4096);
  k_cvt<<<48, 256, 0, stream>>>(W1, wb_W1, 12288);
  k_cvt_pad<<<12, 256, 0, stream>>>(W2, wb_W2);

  k_nodeproj<<<NN / 16, 128, 0, stream>>>(nodes, wb_q_n, wb_k_n, wb_v_n, wb_n_e, q,
                                          kn, vn, ne);
  k_edge_scores<<<NE / 64, 128, 0, stream>>>(edges, src, tgt, q, kn, ne, wb_q_e,
                                             wb_k_e, wb_e_n, s_n, s_e, m_n, m_e,
                                             ebf);
  k_expsum<<<2048, 256, 0, stream>>>(s_n, s_e, src, tgt, m_n, m_e, den_n, den_e);
  k_aggregate<<<NE / 128, 256, 0, stream>>>(ebf, src, tgt, vn, wb_v_e, s_n, s_e,
                                            den_n, den_e, msg, pooled);
  k_outproj<<<NN / 16, 64, 0, stream>>>(nodes, msg, pooled, wb_o_n, wb_o_e, nnb, po);
  k_classifier<<<NE / 128, 256, 0, stream>>>(ebf, src, tgt, nnb, po, wb_W1, b1,
                                             wb_W2, b2, out);
}
